// aGCN_60129542144186
// MI455X (gfx1250) — matrix operand order repaired and checked by reading, not compiled, not run on hardware
//
#include <hip/hip_runtime.h>
#include <cstdint>
#include <cstddef>

// ---------------------------------------------------------------------------
// Problem constants (from reference)
// ---------------------------------------------------------------------------
#define P_DIM   512
#define C_DIM   51
#define N_OBJ   100000
#define N_REL   200000
#define E_SKIP  1600000
#define E_OR    400000
#define E_RO    400000
#define SLOPE   0.2f

typedef __bf16 v16bf __attribute__((ext_vector_type(16)));
typedef float  v8f   __attribute__((ext_vector_type(8)));

#define LDS_STRIDE 520   // halves per W^T column: 512 + 8 pad (bank rotate, 16B aligned)
#define LDS_COLS   52    // 51 real columns + 1 zero dummy column for n in 51..63

// A 16x32 bf16 fragment: lanes 0-15 hold K{0..15}, lanes 16-31 K{16..31}
__device__ __forceinline__ v16bf load_a_frag(const float* __restrict__ xrow,
                                             int kbase, int half)
{
    const float4* p0 = (const float4*)(xrow + kbase + half * 16);
    const float4* p1 = (const float4*)(xrow + kbase + half * 16 + 8);
    float4 x0 = p0[0], x1 = p0[1], x2 = p1[0], x3 = p1[1];
    v16bf a;
    a[0]=(__bf16)x0.x;  a[1]=(__bf16)x0.y;  a[2]=(__bf16)x0.z;  a[3]=(__bf16)x0.w;
    a[4]=(__bf16)x1.x;  a[5]=(__bf16)x1.y;  a[6]=(__bf16)x1.z;  a[7]=(__bf16)x1.w;
    a[8]=(__bf16)x2.x;  a[9]=(__bf16)x2.y;  a[10]=(__bf16)x2.z; a[11]=(__bf16)x2.w;
    a[12]=(__bf16)x3.x; a[13]=(__bf16)x3.y; a[14]=(__bf16)x3.z; a[15]=(__bf16)x3.w;
    return a;
}

// ---------------------------------------------------------------------------
// WMMA projection: H[n_rows,51] = X[n_rows,512] @ W[512,51]  (bf16 in, f32 acc)
// 256 threads = 8 waves; wave (m_sub,n_sub) owns a 16x16 D tile.
// Block covers 32 rows x 64 cols (cols 51..63 read the zeroed dummy column).
// Branch-free, software-pipelined inner loop (prefetch k+1 before WMMA k).
// ---------------------------------------------------------------------------
__global__ __launch_bounds__(256)
void proj_wmma_bf16(const float* __restrict__ X, const float* __restrict__ W,
                    float* __restrict__ H, int n_rows)
{
    __shared__ __bf16 Wt[LDS_COLS * LDS_STRIDE];   // W^T as bf16: Wt[n][k]
    const int tid = threadIdx.x;
    for (int i = tid; i < LDS_COLS * P_DIM; i += 256) {
        int n = i / P_DIM;
        int k = i - n * P_DIM;
        Wt[n * LDS_STRIDE + k] = (n < C_DIM) ? (__bf16)W[k * C_DIM + n] : (__bf16)0.0f;
    }
    __syncthreads();

    const int lane  = tid & 31;
    const int wid   = tid >> 5;
    const int m_sub = wid >> 2;          // 0..1
    const int n_sub = wid & 3;           // 0..3
    const int half  = lane >> 4;         // 0 | 1
    const int l16   = lane & 15;
    const int base  = blockIdx.x * 32 + m_sub * 16;
    const int r     = base + l16;                       // A-matrix row of this lane
    const int rld   = (r < n_rows) ? r : (n_rows - 1);  // clamped load row
    const int ncol  = n_sub * 16 + l16;                 // B/D column of this lane
    const int ncb   = (ncol < C_DIM) ? ncol : C_DIM;    // 51 => zero dummy column

    v8f acc;
#pragma unroll
    for (int i = 0; i < 8; ++i) acc[i] = 0.0f;

    const float* xrow = X + (size_t)rld * P_DIM;
    const __bf16* wcol = &Wt[ncb * LDS_STRIDE + half * 16];

    // Software pipeline: prefetch step k+1's fragments before issuing WMMA k.
    v16bf a_cur = load_a_frag(xrow, 0, half);
    v16bf b_cur = *(const v16bf*)(wcol + 0);

#pragma unroll
    for (int kk = 0; kk < 16; ++kk) {
        const int knext = (kk < 15) ? (kk + 1) * 32 : kk * 32;  // last: harmless reload
        v16bf a_nxt = load_a_frag(xrow, knext, half);
        v16bf b_nxt = *(const v16bf*)(wcol + knext);
        acc = __builtin_amdgcn_wmma_f32_16x16x32_bf16(
                  false, a_cur, false, b_cur, (short)0, acc, false, false);
        a_cur = a_nxt;
        b_cur = b_nxt;
    }

    if (ncol < C_DIM) {
        // D layout: acc[rr] -> (M = rr + 8*half, N = l16)
#pragma unroll
        for (int rr = 0; rr < 8; ++rr) {
            int row = base + rr + 8 * half;
            if (row < n_rows) H[(size_t)row * C_DIM + ncol] = acc[rr];
        }
    }
}

// ---------------------------------------------------------------------------
// e[row] = dot(H[row,:51], a[:51])  -- one wave32 per row
// ---------------------------------------------------------------------------
__global__ __launch_bounds__(256)
void dot_rows_51(const float* __restrict__ H, const float* __restrict__ a,
                 float* __restrict__ e, int n_rows)
{
    const int lane = threadIdx.x & 31;
    const int row  = (blockIdx.x * 256 + threadIdx.x) >> 5;
    if (row >= n_rows) return;
    const float* h = H + (size_t)row * C_DIM;
    float s = h[lane] * a[lane];
    if (lane + 32 < C_DIM) s += h[lane + 32] * a[lane + 32];
#pragma unroll
    for (int off = 16; off > 0; off >>= 1) s += __shfl_xor(s, off, 32);
    if (lane == 0) e[row] = s;
}

// ---------------------------------------------------------------------------
// Monotone float<->uint key (unsigned order == float order)
// ---------------------------------------------------------------------------
__device__ __forceinline__ unsigned fkey(float f) {
    unsigned b = __float_as_uint(f);
    return (b & 0x80000000u) ? ~b : (b | 0x80000000u);
}
__device__ __forceinline__ float finv(unsigned k) {
    unsigned b = (k & 0x80000000u) ? (k ^ 0x80000000u) : ~k;
    return __uint_as_float(b);
}

// Phase 1: e = leaky_relu(e_src[src] + e_dst[dst]); segment max over dst
__global__ __launch_bounds__(256)
void edge_phase1(const float* __restrict__ es, const float* __restrict__ ed,
                 const int* __restrict__ src, const int* __restrict__ dst,
                 int E, unsigned* __restrict__ mkey, float* __restrict__ escr)
{
    int i = blockIdx.x * 256 + threadIdx.x;
    if (i >= E) return;
    float e = es[src[i]] + ed[dst[i]];
    e = (e > 0.0f) ? e : SLOPE * e;
    escr[i] = e;
    atomicMax(&mkey[dst[i]], fkey(e));
}

// Phase 2: ex = exp(e - m[dst]); segment sum over dst
__global__ __launch_bounds__(256)
void edge_phase2(const int* __restrict__ dst, int E,
                 const unsigned* __restrict__ mkey,
                 float* __restrict__ escr, float* __restrict__ denom)
{
    int i = blockIdx.x * 256 + threadIdx.x;
    if (i >= E) return;
    int d = dst[i];
    float ex = __expf(escr[i] - finv(mkey[d]));
    escr[i] = ex;
    atomicAdd(&denom[d], ex);
}

// Phase 3: out[dst,:] += h_src[src,:] * alpha   -- one wave32 per edge
__global__ __launch_bounds__(256)
void edge_phase3(const int* __restrict__ src, const int* __restrict__ dst, int E,
                 const float* __restrict__ escr, const float* __restrict__ denom,
                 const float* __restrict__ Hsrc, float* __restrict__ out)
{
    const int lane = threadIdx.x & 31;
    const int e    = (blockIdx.x * 256 + threadIdx.x) >> 5;
    if (e >= E) return;
    const int s = src[e], d = dst[e];
    const float alpha = escr[e] / denom[d];
    const float* hs = Hsrc + (size_t)s * C_DIM;
    float* o = out + (size_t)d * C_DIM;
    atomicAdd(&o[lane], hs[lane] * alpha);
    if (lane + 32 < C_DIM)
        atomicAdd(&o[lane + 32], hs[lane + 32] * alpha);
}

// obj_out += b_skip + b_ro
__global__ __launch_bounds__(256)
void add_bias2(float* __restrict__ out, const float* __restrict__ b1,
               const float* __restrict__ b2, int n_rows)
{
    size_t n = (size_t)n_rows * C_DIM;
    for (size_t i = blockIdx.x * 256ull + threadIdx.x; i < n; i += (size_t)gridDim.x * 256ull) {
        int c = (int)(i % C_DIM);
        out[i] += b1[c] + b2[c];
    }
}

// rel_probs = softmax(out_rel + b_or, axis=1)  -- one wave32 per row, in place
__global__ __launch_bounds__(256)
void softmax_rows(float* __restrict__ out, const float* __restrict__ bias, int n_rows)
{
    const int lane = threadIdx.x & 31;
    const int row  = (blockIdx.x * 256 + threadIdx.x) >> 5;
    if (row >= n_rows) return;
    float* o = out + (size_t)row * C_DIM;
    const bool has2 = (lane + 32) < C_DIM;
    float v0 = o[lane] + bias[lane];
    float v1 = has2 ? (o[lane + 32] + bias[lane + 32]) : -3.4e38f;
    float mx = fmaxf(v0, v1);
#pragma unroll
    for (int off = 16; off > 0; off >>= 1) mx = fmaxf(mx, __shfl_xor(mx, off, 32));
    float e0 = __expf(v0 - mx);
    float e1 = has2 ? __expf(v1 - mx) : 0.0f;
    float s = e0 + e1;
#pragma unroll
    for (int off = 16; off > 0; off >>= 1) s += __shfl_xor(s, off, 32);
    float inv = 1.0f / s;
    o[lane] = e0 * inv;
    if (has2) o[lane + 32] = e1 * inv;
}

// Zero a word region
__global__ __launch_bounds__(256)
void zero_words(uint32_t* __restrict__ p, size_t n)
{
    for (size_t i = blockIdx.x * 256ull + threadIdx.x; i < n; i += (size_t)gridDim.x * 256ull)
        p[i] = 0u;
}

// ---------------------------------------------------------------------------
// Launch
// ---------------------------------------------------------------------------
extern "C" void kernel_launch(void* const* d_in, const int* in_sizes, int n_in,
                              void* d_out, int out_size, void* d_ws, size_t ws_size,
                              hipStream_t stream)
{
    (void)in_sizes; (void)n_in; (void)out_size; (void)ws_size;

    const float* x_obj     = (const float*)d_in[0];
    const float* x_rel     = (const float*)d_in[1];
    const float* W_skip    = (const float*)d_in[2];
    const float* a_src_sk  = (const float*)d_in[3];
    const float* a_dst_sk  = (const float*)d_in[4];
    const float* b_skip    = (const float*)d_in[5];
    const float* W_or_src  = (const float*)d_in[6];
    const float* W_or_dst  = (const float*)d_in[7];
    const float* a_src_or  = (const float*)d_in[8];
    const float* a_dst_or  = (const float*)d_in[9];
    const float* b_or      = (const float*)d_in[10];
    const float* W_ro_src  = (const float*)d_in[11];
    const float* W_ro_dst  = (const float*)d_in[12];
    const float* a_src_ro  = (const float*)d_in[13];
    const float* a_dst_ro  = (const float*)d_in[14];
    const float* b_ro      = (const float*)d_in[15];
    const int*   skip_src  = (const int*)d_in[16];
    const int*   skip_dst  = (const int*)d_in[17];
    const int*   or_src    = (const int*)d_in[18];
    const int*   or_dst    = (const int*)d_in[19];
    const int*   ro_src    = (const int*)d_in[20];
    const int*   ro_dst    = (const int*)d_in[21];

    float* out_rel = (float*)d_out;                      // [N_REL, 51]
    float* out_obj = out_rel + (size_t)N_REL * C_DIM;    // [N_OBJ, 51]

    // ----- workspace partition (floats) -----
    float* ws = (float*)d_ws;
    size_t o = 0;
    float* h_skip = ws + o; o += (size_t)N_OBJ * C_DIM;
    float* h_or_s = ws + o; o += (size_t)N_OBJ * C_DIM;
    float* h_or_d = ws + o; o += (size_t)N_REL * C_DIM;
    float* h_ro_s = ws + o; o += (size_t)N_REL * C_DIM;
    float* h_ro_d = ws + o; o += (size_t)N_OBJ * C_DIM;
    float* e_s_sk = ws + o; o += N_OBJ;
    float* e_d_sk = ws + o; o += N_OBJ;
    float* e_s_or = ws + o; o += N_OBJ;
    float* e_d_or = ws + o; o += N_REL;
    float* e_s_ro = ws + o; o += N_REL;
    float* e_d_ro = ws + o; o += N_OBJ;
    float* stats  = ws + o;                              // mkey+denom, zeroed as block
    unsigned* mk_sk = (unsigned*)(ws + o); o += N_OBJ;
    float*    dn_sk = ws + o;              o += N_OBJ;
    unsigned* mk_or = (unsigned*)(ws + o); o += N_REL;
    float*    dn_or = ws + o;              o += N_REL;
    unsigned* mk_ro = (unsigned*)(ws + o); o += N_OBJ;
    float*    dn_ro = ws + o;              o += N_OBJ;
    size_t stats_words = (size_t)2 * (N_OBJ + N_REL + N_OBJ);
    float* es_sk = ws + o; o += E_SKIP;
    float* es_or = ws + o; o += E_OR;
    float* es_ro = ws + o; o += E_RO;

    // ----- zero accumulators (d_out is poisoned; graphs replay this) -----
    size_t out_words = (size_t)(N_REL + N_OBJ) * C_DIM;
    zero_words<<<2048, 256, 0, stream>>>((uint32_t*)d_out, out_words);
    zero_words<<<1024, 256, 0, stream>>>((uint32_t*)stats, stats_words);

    // ----- projections (WMMA) -----
    proj_wmma_bf16<<<(N_OBJ + 31) / 32, 256, 0, stream>>>(x_obj, W_skip,   h_skip, N_OBJ);
    proj_wmma_bf16<<<(N_OBJ + 31) / 32, 256, 0, stream>>>(x_obj, W_or_src, h_or_s, N_OBJ);
    proj_wmma_bf16<<<(N_REL + 31) / 32, 256, 0, stream>>>(x_rel, W_or_dst, h_or_d, N_REL);
    proj_wmma_bf16<<<(N_REL + 31) / 32, 256, 0, stream>>>(x_rel, W_ro_src, h_ro_s, N_REL);
    proj_wmma_bf16<<<(N_OBJ + 31) / 32, 256, 0, stream>>>(x_obj, W_ro_dst, h_ro_d, N_OBJ);

    // ----- attention logits per node -----
    dot_rows_51<<<(N_OBJ + 7) / 8, 256, 0, stream>>>(h_skip, a_src_sk, e_s_sk, N_OBJ);
    dot_rows_51<<<(N_OBJ + 7) / 8, 256, 0, stream>>>(h_skip, a_dst_sk, e_d_sk, N_OBJ);
    dot_rows_51<<<(N_OBJ + 7) / 8, 256, 0, stream>>>(h_or_s, a_src_or, e_s_or, N_OBJ);
    dot_rows_51<<<(N_REL + 7) / 8, 256, 0, stream>>>(h_or_d, a_dst_or, e_d_or, N_REL);
    dot_rows_51<<<(N_REL + 7) / 8, 256, 0, stream>>>(h_ro_s, a_src_ro, e_s_ro, N_REL);
    dot_rows_51<<<(N_OBJ + 7) / 8, 256, 0, stream>>>(h_ro_d, a_dst_ro, e_d_ro, N_OBJ);

    // ----- segment softmax + scatter: skip (obj->obj), or (obj->rel), ro (rel->obj) -----
    edge_phase1<<<(E_SKIP + 255) / 256, 256, 0, stream>>>(e_s_sk, e_d_sk, skip_src, skip_dst, E_SKIP, mk_sk, es_sk);
    edge_phase1<<<(E_OR   + 255) / 256, 256, 0, stream>>>(e_s_or, e_d_or, or_src,   or_dst,   E_OR,   mk_or, es_or);
    edge_phase1<<<(E_RO   + 255) / 256, 256, 0, stream>>>(e_s_ro, e_d_ro, ro_src,   ro_dst,   E_RO,   mk_ro, es_ro);

    edge_phase2<<<(E_SKIP + 255) / 256, 256, 0, stream>>>(skip_dst, E_SKIP, mk_sk, es_sk, dn_sk);
    edge_phase2<<<(E_OR   + 255) / 256, 256, 0, stream>>>(or_dst,   E_OR,   mk_or, es_or, dn_or);
    edge_phase2<<<(E_RO   + 255) / 256, 256, 0, stream>>>(ro_dst,   E_RO,   mk_ro, es_ro, dn_ro);

    edge_phase3<<<(E_SKIP + 7) / 8, 256, 0, stream>>>(skip_src, skip_dst, E_SKIP, es_sk, dn_sk, h_skip, out_obj);
    edge_phase3<<<(E_OR   + 7) / 8, 256, 0, stream>>>(or_src,   or_dst,   E_OR,   es_or, dn_or, h_or_s, out_rel);
    edge_phase3<<<(E_RO   + 7) / 8, 256, 0, stream>>>(ro_src,   ro_dst,   E_RO,   es_ro, dn_ro, h_ro_s, out_obj);

    // ----- finalize -----
    add_bias2<<<2048, 256, 0, stream>>>(out_obj, b_skip, b_ro, N_OBJ);
    softmax_rows<<<(N_REL + 7) / 8, 256, 0, stream>>>(out_rel, b_or, N_REL);
}
